// Attention_16990890623172
// MI455X (gfx1250) — matrix operand order repaired and checked by reading, not compiled, not run on hardware
//
#include <hip/hip_runtime.h>
#include <hip/hip_bf16.h>
#include <math.h>

// ---------------------------------------------------------------------------
// Bahdanau attention, MI455X (gfx1250, wave32, WMMA).
//   scores[m] = v . tanh( E[m,:] @ We^T + proj_h[b(m),:] ),  m = b*S+s
//   attn = softmax_s(scores);  context[b,:] = attn[b,:] @ E[b,:,:]
// Big GEMM: M=65536, N=1024, K=1024 -> bf16 WMMA 16x16x32, f32 accum.
// ---------------------------------------------------------------------------

typedef __attribute__((ext_vector_type(16))) __bf16 v16bf;
typedef __attribute__((ext_vector_type(8)))  float  v8f;

#define HH 1024
#define BB 32
#define SS 2048
#define MT (BB*SS)          // 65536 rows total

// ---------------------------------------------------------------------------
// Kernel 1: pack We (= W_attn[:, H:2H], row-major [o,k]) into bf16 WMMA
// B-fragment layout.  Fragment (ntile g, kstep kt) = 32 lanes x 16 bf16:
//   lane L: n = g*16 + (L&15),  k = kt*32 + (L>>4)*16 + e,  e = 0..15
// Slot index = (g*32 + kt)*32 + L  == flat thread id (65536 threads).
// ---------------------------------------------------------------------------
__global__ void pack_B_kernel(const float* __restrict__ W, v16bf* __restrict__ Bp)
{
    int t   = blockIdx.x * blockDim.x + threadIdx.x;   // 0..65535
    int g   = t >> 10;                                 // ntile 0..63
    int rem = t & 1023;
    int kt  = rem >> 5;                                // kstep 0..31
    int l   = rem & 31;                                // lane
    int n   = (g << 4) + (l & 15);
    int k   = (kt << 5) + ((l >> 4) << 4);
    const float* src = W + (size_t)n * 2048 + 1024 + k;
    v16bf out;
#pragma unroll
    for (int e = 0; e < 16; ++e) out[e] = (__bf16)src[e];
    Bp[t] = out;
}

// ---------------------------------------------------------------------------
// Kernel 2: proj_h[b,o] = hidden[-1][b,:] . W_attn[o, 0:H] + b_attn[o]
// 32768 outputs, trivial FLOPs -> scalar f32 with float4 loads.
// ---------------------------------------------------------------------------
__global__ void proj_h_kernel(const float* __restrict__ hidden,
                              const float* __restrict__ W,
                              const float* __restrict__ b_attn,
                              float* __restrict__ ph)
{
    int t = blockIdx.x * blockDim.x + threadIdx.x;     // 0..32767
    int b = t >> 10, o = t & 1023;
    const float4* hr = (const float4*)(hidden + (size_t)(BB + b) * HH); // layer -1
    const float4* wr = (const float4*)(W + (size_t)o * 2048);
    float acc = 0.f;
#pragma unroll 4
    for (int i = 0; i < 256; ++i) {
        float4 a = hr[i], w = wr[i];
        acc += a.x * w.x + a.y * w.y + a.z * w.z + a.w * w.w;
    }
    ph[t] = acc + b_attn[o];
}

// ---------------------------------------------------------------------------
// Kernel 3: tiled GEMM + fused tanh/v-dot epilogue.
// Grid (512, 8): blockIdx.x -> 128-row M-tile, blockIdx.y -> 128-col N-tile.
// 8 waves: wave w owns rows [m0+16w, m0+16w+16) x all 8 N-subtiles.
// K chunked by 128; A chunk staged f32->bf16 into LDS in A-fragment layout:
//   lane L<16 : row M=L,    elems 0..15 = K kt*32+0..15
//   lane L>=16: row M=L-16, elems 0..15 = K kt*32+16..31
// ---------------------------------------------------------------------------
__global__ __launch_bounds__(256) void scores_kernel(
    const float* __restrict__ E, const v16bf* __restrict__ Bp,
    const float* __restrict__ ph, const float* __restrict__ v,
    float* __restrict__ scores)
{
    __shared__ v16bf Ash[1024];            // 32 frags * 32 lanes = 32 KB
    const int m0   = blockIdx.x * 128;
    const int n0   = blockIdx.y * 128;
    const int tid  = threadIdx.x;
    const int w    = tid >> 5;             // wave id 0..7 (M strip)
    const int lane = tid & 31;
    const int bb   = m0 >> 11;             // batch (tiles never cross: 2048%128==0)

    v8f acc[8];
    const v8f vzero = {0.f,0.f,0.f,0.f,0.f,0.f,0.f,0.f};
#pragma unroll
    for (int i = 0; i < 8; ++i) acc[i] = vzero;

    for (int kc = 0; kc < 8; ++kc) {       // K chunks of 128
        __syncthreads();
        // ---- stage A chunk: 1024 (frag,lane) slots, 4 per thread ----
#pragma unroll
        for (int it = 0; it < 4; ++it) {
            int slot = tid + (it << 8);    // 0..1023
            int f  = slot >> 5;            // frag: strip s = f>>2, kstep t = f&3
            int l  = slot & 31;
            int s  = f >> 2, t = f & 3;
            int row = m0 + (s << 4) + (l & 15);
            int kb  = (kc << 7) + (t << 5) + ((l >> 4) << 4);
            const float* p = E + (size_t)row * HH + kb;
            float4 f0 = *(const float4*)(p);
            float4 f1 = *(const float4*)(p + 4);
            float4 f2 = *(const float4*)(p + 8);
            float4 f3 = *(const float4*)(p + 12);
            v16bf a;
            a[0]=(__bf16)f0.x; a[1]=(__bf16)f0.y; a[2]=(__bf16)f0.z; a[3]=(__bf16)f0.w;
            a[4]=(__bf16)f1.x; a[5]=(__bf16)f1.y; a[6]=(__bf16)f1.z; a[7]=(__bf16)f1.w;
            a[8]=(__bf16)f2.x; a[9]=(__bf16)f2.y; a[10]=(__bf16)f2.z; a[11]=(__bf16)f2.w;
            a[12]=(__bf16)f3.x; a[13]=(__bf16)f3.y; a[14]=(__bf16)f3.z; a[15]=(__bf16)f3.w;
            Ash[slot] = a;
        }
        __syncthreads();
        // ---- compute: 4 ksteps x 8 N-tiles of WMMA ----
#pragma unroll
        for (int t = 0; t < 4; ++t) {
            v16bf af = Ash[((w << 2) + t) * 32 + lane];
#pragma unroll
            for (int nt = 0; nt < 8; ++nt) {
                int g = (n0 >> 4) + nt;
                v16bf bf = Bp[(size_t)(((g << 5) + (kc << 2) + t) * 32 + lane)];
                acc[nt] = __builtin_amdgcn_wmma_f32_16x16x32_bf16(
                    false, af, false, bf, (short)0, acc[nt], false, false);
            }
        }
    }

    // ---- epilogue: tanh, dot with v, reduce across 16 N-lanes ----
    // C layout: VGPR r -> row m0+16w + r + 8*(lane>>4); col n0 + nt*16 + (lane&15)
    float part[8];
#pragma unroll
    for (int r = 0; r < 8; ++r) part[r] = 0.f;
    const int nb = lane & 15, hh = lane >> 4;
#pragma unroll
    for (int nt = 0; nt < 8; ++nt) {
        int n = n0 + (nt << 4) + nb;
        float phv = ph[(bb << 10) + n];
        float vv  = v[n];
#pragma unroll
        for (int r = 0; r < 8; ++r)
            part[r] += tanhf(acc[nt][r] + phv) * vv;
    }
#pragma unroll
    for (int r = 0; r < 8; ++r) {
        float x = part[r];
        x += __shfl_xor(x, 1, 32);
        x += __shfl_xor(x, 2, 32);
        x += __shfl_xor(x, 4, 32);
        x += __shfl_xor(x, 8, 32);
        if (nb == 0)
            atomicAdd(&scores[m0 + (w << 4) + (hh << 3) + r], x);
    }
}

// ---------------------------------------------------------------------------
// Kernel 4: softmax over S per batch row (one 256-thread block per b).
// ---------------------------------------------------------------------------
__global__ void softmax_kernel(const float* __restrict__ scores,
                               float* __restrict__ attn)
{
    __shared__ float red[256];
    int b = blockIdx.x, tid = threadIdx.x;
    const float* sr = scores + (b << 11);
    float e[8];
    float mx = -1e30f;
#pragma unroll
    for (int i = 0; i < 8; ++i) { e[i] = sr[tid + (i << 8)]; mx = fmaxf(mx, e[i]); }
    red[tid] = mx; __syncthreads();
    for (int o = 128; o > 0; o >>= 1) {
        if (tid < o) red[tid] = fmaxf(red[tid], red[tid + o]);
        __syncthreads();
    }
    mx = red[0]; __syncthreads();
    float s = 0.f;
#pragma unroll
    for (int i = 0; i < 8; ++i) { e[i] = expf(e[i] - mx); s += e[i]; }
    red[tid] = s; __syncthreads();
    for (int o = 128; o > 0; o >>= 1) {
        if (tid < o) red[tid] += red[tid + o];
        __syncthreads();
    }
    float inv = 1.f / red[0];
#pragma unroll
    for (int i = 0; i < 8; ++i) attn[(b << 11) + tid + (i << 8)] = e[i] * inv;
}

// ---------------------------------------------------------------------------
// Kernel 5: context[b,h] = sum_s attn[b,s] * E[b,s,h]  (bandwidth bound).
// Grid (32, 4), 256 threads; attn chunk staged through LDS for broadcast.
// ---------------------------------------------------------------------------
__global__ void context_kernel(const float* __restrict__ E,
                               const float* __restrict__ attn,
                               float* __restrict__ ctx)
{
    __shared__ float aw[256];
    int b = blockIdx.x;
    int h = blockIdx.y * 256 + threadIdx.x;
    float acc = 0.f;
    for (int s0 = 0; s0 < SS; s0 += 256) {
        __syncthreads();
        aw[threadIdx.x] = attn[(b << 11) + s0 + threadIdx.x];
        __syncthreads();
        const float* ep = E + ((size_t)((b << 11) + s0) << 10) + h;
#pragma unroll 8
        for (int j = 0; j < 256; ++j) acc += aw[j] * ep[(size_t)j << 10];
    }
    ctx[(b << 10) + h] = acc;
}

// ---------------------------------------------------------------------------
// Host launcher.  ws: [ph 32768 f32][scores 65536 f32][Bpack 65536 x v16bf]
// d_out: [context 32768 f32][attn 65536 f32]
// ---------------------------------------------------------------------------
extern "C" void kernel_launch(void* const* d_in, const int* in_sizes, int n_in,
                              void* d_out, int out_size, void* d_ws, size_t ws_size,
                              hipStream_t stream)
{
    const float* hidden = (const float*)d_in[0];
    const float* E      = (const float*)d_in[1];
    const float* W      = (const float*)d_in[2];
    const float* b_attn = (const float*)d_in[3];
    const float* v      = (const float*)d_in[4];

    float* out  = (float*)d_out;
    float* ctx  = out;              // 32*1024
    float* attn = out + 32768;      // 32*2048

    float* ph     = (float*)d_ws;
    float* scores = ph + 32768;
    v16bf* Bp     = (v16bf*)((char*)d_ws + (size_t)(32768 + 65536) * sizeof(float));

    hipMemsetAsync(scores, 0, (size_t)65536 * sizeof(float), stream);

    pack_B_kernel<<<256, 256, 0, stream>>>(W, Bp);
    proj_h_kernel<<<128, 256, 0, stream>>>(hidden, W, b_attn, ph);

    dim3 g3(512, 8);
    scores_kernel<<<g3, 256, 0, stream>>>(E, Bp, ph, v, scores);

    softmax_kernel<<<32, 256, 0, stream>>>(scores, attn);

    dim3 g5(32, 4);
    context_kernel<<<g5, 256, 0, stream>>>(E, attn, ctx);
}
